// CrossAttention_12103217840349
// MI455X (gfx1250) — matrix operand order repaired and checked by reading, not compiled, not run on hardware
//
#include <hip/hip_runtime.h>
#include <hip/hip_bf16.h>
#include <math.h>
#include <stddef.h>

// ---------------------------------------------------------------------------
// CrossAttention for MI455X (gfx1250, wave32, WMMA + TDM).
// bf16 storage / f32 accumulation via v_wmma_f32_16x16x32_bf16.
// alibi (268 MB) is the dominant HBM traffic: read once, coalesced, prefetched.
// K tiles staged by the Tensor Data Mover; softmax row-sum computed by a
// ones-column WMMA; row-max reduced with v_permlane16 (VALU, not LDS pipe).
// ---------------------------------------------------------------------------

typedef __bf16 bf16;
typedef __attribute__((ext_vector_type(16))) __bf16 bf16x16;
typedef __attribute__((ext_vector_type(8)))  __bf16 bf16x8;
typedef __attribute__((ext_vector_type(8)))  float  f32x8;
typedef __attribute__((ext_vector_type(4)))  unsigned int u32x4;
typedef __attribute__((ext_vector_type(8)))  int  i32x8;
typedef __attribute__((ext_vector_type(4)))  int  i32x4;

#define HEADS   16
#define DH      64
#define NSEQ    2048
#define DMODEL  1024
#define BATCH   2
#define MROWS   (BATCH * NSEQ)   // 4096

__device__ __forceinline__ f32x8 wmma_bf16(bf16x16 a, bf16x16 b, f32x8 c) {
    // 8 args: (neg_a, A, neg_b, B, c_mod, C, reuse_a, reuse_b)
    return __builtin_amdgcn_wmma_f32_16x16x32_bf16(false, a, false, b,
                                                   (short)0, c, false, false);
}

__device__ __forceinline__ f32x8 zero8() {
    f32x8 z; for (int i = 0; i < 8; ++i) z[i] = 0.f; return z;
}

// xor-butterfly max within each 16-lane row via v_permlane16_b32 (VALU pipe).
__device__ __forceinline__ float permx_max(float v, int s0, int s1) {
    const int iv = __builtin_bit_cast(int, v);
    const int pv = __builtin_amdgcn_permlane16(iv, iv, s0, s1, false, false);
    return fmaxf(v, __builtin_bit_cast(float, pv));
}

// ------------------------------- fp32 -> bf16 ------------------------------
__global__ __launch_bounds__(256)
void cvt_bf16_kernel(const float* __restrict__ in, bf16* __restrict__ out, int n) {
    int i = blockIdx.x * 256 + threadIdx.x;
    if (i < n) out[i] = (bf16)in[i];
}

// --------------------------- LayerNorm + bf16 cast -------------------------
__global__ __launch_bounds__(256)
void ln_bf16_kernel(const float* __restrict__ x, const float* __restrict__ w,
                    const float* __restrict__ b, bf16* __restrict__ out) {
    const int row = blockIdx.x;
    const int tid = threadIdx.x;
    const float* xr = x + (size_t)row * DMODEL;

    float v[4], s = 0.f, ss = 0.f;
    for (int i = 0; i < 4; ++i) {
        v[i] = xr[tid + i * 256];
        s += v[i]; ss += v[i] * v[i];
    }
    for (int m = 16; m >= 1; m >>= 1) {
        s  += __shfl_xor(s, m, 32);
        ss += __shfl_xor(ss, m, 32);
    }
    __shared__ float sw[8], ssw[8];
    if ((tid & 31) == 0) { sw[tid >> 5] = s; ssw[tid >> 5] = ss; }
    __syncthreads();
    s = 0.f; ss = 0.f;
    for (int i = 0; i < 8; ++i) { s += sw[i]; ss += ssw[i]; }

    const float mu   = s  * (1.f / DMODEL);
    const float var  = ss * (1.f / DMODEL) - mu * mu;
    const float rstd = rsqrtf(var + 1e-5f);
    for (int i = 0; i < 4; ++i) {
        int c = tid + i * 256;
        out[(size_t)row * DMODEL + c] = (bf16)((v[i] - mu) * rstd * w[c] + b[c]);
    }
}

// ------------------------------ WMMA GEMM ----------------------------------
// C[M,N] = A[M,K] x W[N,K]^T. 128 threads = 4 waves (2x2), wave tile 64x64
// (4x4 WMMA subtiles -> 16 WMMAs per 16 b128 loads per K-step), WG tile 128x128.
// Templated on output mode so each epilogue is straight-line (no spills).
template <bool WRITE_F32>
__global__ __launch_bounds__(128, 1)
void gemm_wmma_kernel(const bf16* __restrict__ A, const bf16* __restrict__ W,
                      bf16* __restrict__ Cbf, float* __restrict__ Cf,
                      const float* __restrict__ bias, int M, int N, int K) {
    const int lane = threadIdx.x & 31;
    const int ll = lane & 15, lh = lane >> 4;
    const int wave = threadIdx.x >> 5;     // 0..3
    const int wm = wave & 1, wn = wave >> 1;
    const int mbase = blockIdx.y * 128 + wm * 64;
    const int nbase = blockIdx.x * 128 + wn * 64;

    f32x8 acc[4][4];
    for (int i = 0; i < 4; ++i)
        for (int j = 0; j < 4; ++j) acc[i][j] = zero8();

    for (int kt = 0; kt < K; kt += 32) {
        // B fragment: N = lane&15, K element = e + 16*lane_hi (one 32B load)
        bf16x16 bfr[4];
        for (int j = 0; j < 4; ++j) {
            const int n = nbase + j * 16 + ll;
            bfr[j] = *(const bf16x16*)(W + (size_t)n * K + kt + 16 * lh);
        }
        // A fragment: M = lane&15, K chunks at 16*hi and 16*hi+8 (two 16B loads)
        bf16x16 afr[4];
        for (int i = 0; i < 4; ++i) {
            const int m = mbase + i * 16 + ll;
            const bf16* ar = A + (size_t)m * K + kt;
            union { bf16x16 v; bf16x8 h[2]; } u;
            u.h[0] = *(const bf16x8*)(ar + 16 * lh);
            u.h[1] = *(const bf16x8*)(ar + 16 * lh + 8);
            afr[i] = u.v;
        }
        for (int i = 0; i < 4; ++i)
            for (int j = 0; j < 4; ++j)
                acc[i][j] = wmma_bf16(afr[i], bfr[j], acc[i][j]);
    }

    // C/D layout: VGPR r -> row r + 8*lane_hi, col = lane&15
    for (int i = 0; i < 4; ++i)
        for (int j = 0; j < 4; ++j) {
            const int col = nbase + j * 16 + ll;
            if constexpr (WRITE_F32) {
                const float bv = bias[col];
                for (int r = 0; r < 8; ++r) {
                    const int row = mbase + i * 16 + r + 8 * lh;
                    Cf[(size_t)row * N + col] = acc[i][j][r] + bv;
                }
            } else {
                for (int r = 0; r < 8; ++r) {
                    const int row = mbase + i * 16 + r + 8 * lh;
                    Cbf[(size_t)row * N + col] = (bf16)acc[i][j][r];
                }
            }
        }
}

// ---------------------- TDM descriptor for the K tile ----------------------
// 2D tile: 32 rows (keys, stride DMODEL bf16) x 64 cols (dh, contiguous),
// data_size=1 (2 bytes), D#.type=2, lds_addr = byte offset in workgroup LDS.
// amdgpu-toolchain (clang-23) 6-arg builtin:
//   (u32x4 g0, i32x8 g1, i32x4 g2, i32x4 g3, i32x8 g4, i32 cpol)
__device__ __forceinline__ void tdm_load_k_tile(const bf16* gaddr, unsigned lds_off) {
    const unsigned long long ga = (unsigned long long)(size_t)gaddr;
    u32x4 g0;
    g0[0] = 1u;                                        // count=1 user descriptor
    g0[1] = lds_off;                                   // lds_addr (bytes)
    g0[2] = (unsigned)(ga & 0xFFFFFFFFull);            // global_addr[31:0]
    g0[3] = (unsigned)((ga >> 32) & 0x01FFFFFFull)     // global_addr[56:32]
          | (2u << 30);                                // type = 2 ("image")
    i32x8 g1;
    g1[0] = 0x00010000;                                // data_size=1 (2B/elt)
    g1[1] = (int)(64u << 16);                          // tensor_dim0 = 64
    g1[2] = (int)(32u << 16);                          // tensor_dim1 = 32
    g1[3] = (int)(64u << 16);                          // tile_dim0   = 64
    g1[4] = 32;                                        // tile_dim1   = 32
    g1[5] = DMODEL;                                    // tensor_dim0_stride
    g1[6] = 0;
    g1[7] = 0;
    i32x4 z4; z4[0] = z4[1] = z4[2] = z4[3] = 0;       // groups 2/3 unused (2D)
    i32x8 z8; for (int i = 0; i < 8; ++i) z8[i] = 0;   // group 4 unused
    __builtin_amdgcn_tensor_load_to_lds(g0, g1, z4, z4, z8, 0);
}

// ------------------------- Flash attention + alibi -------------------------
// Grid: (n/64, b*h). 128 threads = 4 waves; each wave owns 16 query rows.
// Per 32-key block: TDM stages K (32x64) into LDS while threads transpose V;
// 4 WMMAs for S = Q K^T, online softmax (max via permlane16, sum via a
// ones-column WMMA), P relayout via per-wave LDS, 4 WMMAs for O += P V.
struct FlashSmem {
    alignas(16) bf16 Kblk[32 * 64];       // [key][dim]      offset 0 (TDM dest)
    alignas(16) bf16 Vt[64 * 40];         // [dim][key] padded stride
    alignas(16) bf16 Pst[4][16 * 32];     // per-wave P relayout
};

__global__ __launch_bounds__(128)
void flash_attn_kernel(const bf16* __restrict__ Q, const bf16* __restrict__ Kb,
                       const bf16* __restrict__ V, const float* __restrict__ alibi,
                       bf16* __restrict__ O) {
    const int lane = threadIdx.x & 31;
    const int ll = lane & 15, lh = lane >> 4;
    const int wv = threadIdx.x >> 5;          // 0..3
    const int bh = blockIdx.y;                // 0..31
    const int bb = bh >> 4, h = bh & 15;
    const int q0 = blockIdx.x * 64;
    const int qw = q0 + wv * 16;              // wave's query base

    __shared__ FlashSmem sm;
    const float scale = 0.125f;               // 1/sqrt(64)

    // Q fragments (dh=64 -> two K=32 steps), loaded once
    bf16x16 qf[2];
    {
        const bf16* qrow = Q + ((size_t)(bb * NSEQ + qw + ll)) * DMODEL + h * DH;
        for (int kb = 0; kb < 2; ++kb) {
            union { bf16x16 v; bf16x8 hh[2]; } u;
            u.hh[0] = *(const bf16x8*)(qrow + kb * 32 + 16 * lh);
            u.hh[1] = *(const bf16x8*)(qrow + kb * 32 + 16 * lh + 8);
            qf[kb] = u.v;
        }
    }

    // ones B-fragment: P x ones accumulates the softmax denominator (o[4])
    bf16x16 ones16;
    for (int i = 0; i < 16; ++i) ones16[i] = (bf16)1.0f;

    f32x8 o[5];                               // o[0..3] = O columns, o[4] = l
    for (int s = 0; s < 5; ++s) o[s] = zero8();
    float mi[8];
    for (int r = 0; r < 8; ++r) mi[r] = -INFINITY;

    const int srow = threadIdx.x >> 2;          // 0..31 key row for staging
    const int scol = (threadIdx.x & 3) * 16;    // 0/16/32/48 dim chunk

    for (int j = 0; j < NSEQ; j += 32) {
        // --- TDM: DMA the K tile into LDS (wave 0 issues; EXEC-independent) ---
        if (wv == 0) {
            tdm_load_k_tile(Kb + ((size_t)(bb * NSEQ + j)) * DMODEL + h * DH,
                            (unsigned)offsetof(FlashSmem, Kblk));
        }
        // --- overlap: cooperative transposed staging of the V block ---
        {
            const size_t g = ((size_t)(bb * NSEQ + j + srow)) * DMODEL + h * DH + scol;
            bf16x16 vvv = *(const bf16x16*)(V + g);
            for (int i = 0; i < 16; ++i) sm.Vt[(scol + i) * 40 + srow] = vvv[i];
        }
        // prefetch the next alibi tile (dominant HBM stream)
        if (j + 32 < NSEQ) {
            const size_t pa = ((size_t)h * NSEQ + (qw + ll)) * NSEQ + (j + 32);
            __builtin_prefetch(alibi + pa, 0, 1);
        }
        if (wv == 0) __builtin_amdgcn_s_wait_tensorcnt(0);
        __syncthreads();

        // --- S = Q K^T (two 16-key column tiles) ---
        f32x8 S[2];
        for (int t = 0; t < 2; ++t) {
            f32x8 sacc = zero8();
            for (int kb = 0; kb < 2; ++kb) {
                bf16x16 kf = *(const bf16x16*)(&sm.Kblk[(t * 16 + ll) * 64 + kb * 32 + 16 * lh]);
                sacc = wmma_bf16(qf[kb], kf, sacc);
            }
            S[t] = sacc;
        }

        // --- scale + alibi + online softmax (rows = r + 8*lh) ---
        float p0[8], p1[8], rm[8];
        for (int r = 0; r < 8; ++r) {
            const size_t arow = ((size_t)h * NSEQ + (qw + r + 8 * lh)) * NSEQ + j + ll;
            p0[r] = S[0][r] * scale + alibi[arow];
            p1[r] = S[1][r] * scale + alibi[arow + 16];
            rm[r] = fmaxf(p0[r], p1[r]);
        }
        // xor-butterfly max over the 16 lanes of each half (VALU permlane16)
        for (int r = 0; r < 8; ++r) {
            rm[r] = permx_max(rm[r], (int)0x67452301, (int)0xEFCDAB89);  // ^1
            rm[r] = permx_max(rm[r], (int)0x54761032, (int)0xDCFE98BA);  // ^2
            rm[r] = permx_max(rm[r], (int)0x32107654, (int)0xBA98FEDC);  // ^4
            rm[r] = permx_max(rm[r], (int)0xFEDCBA98, (int)0x76543210);  // ^8
        }
        for (int r = 0; r < 8; ++r) {
            const float mn = fmaxf(mi[r], rm[r]);
            const float al = __expf(mi[r] - mn);
            p0[r] = __expf(p0[r] - mn);
            p1[r] = __expf(p1[r] - mn);
            mi[r] = mn;
            for (int s = 0; s < 5; ++s) o[s][r] *= al;   // rescale O and l
        }

        // --- relayout P (C-tile layout -> A-fragment) via per-wave LDS ---
        for (int r = 0; r < 8; ++r) {
            sm.Pst[wv][(r + 8 * lh) * 32 + ll]      = (bf16)p0[r];
            sm.Pst[wv][(r + 8 * lh) * 32 + 16 + ll] = (bf16)p1[r];
        }
        asm volatile("s_wait_dscnt 0" ::: "memory");   // wave-local RAW on LDS
        bf16x16 pf;
        {
            union { bf16x16 v; bf16x8 hh[2]; } u;
            u.hh[0] = *(const bf16x8*)(&sm.Pst[wv][ll * 32 + 16 * lh]);
            u.hh[1] = *(const bf16x8*)(&sm.Pst[wv][ll * 32 + 16 * lh + 8]);
            pf = u.v;
        }

        // --- O += P V (dh = 4 tiles of 16 columns) + l += P x ones ---
        for (int s = 0; s < 4; ++s) {
            bf16x16 vf = *(const bf16x16*)(&sm.Vt[(s * 16 + ll) * 40 + 16 * lh]);
            o[s] = wmma_bf16(pf, vf, o[s]);
        }
        o[4] = wmma_bf16(pf, ones16, o[4]);
        __syncthreads();   // protect K/V staging buffers for next block
    }

    // --- epilogue: normalize and store as bf16 [b,n,h*dh] ---
    for (int s = 0; s < 4; ++s)
        for (int r = 0; r < 8; ++r) {
            const int row = bb * NSEQ + qw + r + 8 * lh;
            O[(size_t)row * DMODEL + h * DH + s * 16 + ll] = (bf16)(o[s][r] / o[4][r]);
        }
}

// ------------------------------ launcher -----------------------------------
extern "C" void kernel_launch(void* const* d_in, const int* in_sizes, int n_in,
                              void* d_out, int out_size, void* d_ws, size_t ws_size,
                              hipStream_t stream) {
    (void)in_sizes; (void)n_in; (void)out_size; (void)ws_size;
    const float* x     = (const float*)d_in[0];
    const float* ctx   = (const float*)d_in[1];
    const float* alibi = (const float*)d_in[2];
    const float* Wq    = (const float*)d_in[3];
    const float* Wk    = (const float*)d_in[4];
    const float* Wv    = (const float*)d_in[5];
    const float* Wo    = (const float*)d_in[6];
    const float* bo    = (const float*)d_in[7];
    const float* lnw   = (const float*)d_in[8];
    const float* lnb   = (const float*)d_in[9];
    float* out = (float*)d_out;

    // Workspace (bf16): xn, cn, Q, K, V, attn (MROWS*DMODEL each)
    // + 4 weight matrices (DMODEL*DMODEL each). Total ~56 MiB.
    bf16* xn  = (bf16*)d_ws;
    bf16* cn  = xn  + (size_t)MROWS * DMODEL;
    bf16* Qb  = cn  + (size_t)MROWS * DMODEL;
    bf16* Kb  = Qb  + (size_t)MROWS * DMODEL;
    bf16* Vb  = Kb  + (size_t)MROWS * DMODEL;
    bf16* At  = Vb  + (size_t)MROWS * DMODEL;
    bf16* wqb = At  + (size_t)MROWS * DMODEL;
    bf16* wkb = wqb + (size_t)DMODEL * DMODEL;
    bf16* wvb = wkb + (size_t)DMODEL * DMODEL;
    bf16* wob = wvb + (size_t)DMODEL * DMODEL;

    const int nw = DMODEL * DMODEL;
    cvt_bf16_kernel<<<nw / 256, 256, 0, stream>>>(Wq, wqb, nw);
    cvt_bf16_kernel<<<nw / 256, 256, 0, stream>>>(Wk, wkb, nw);
    cvt_bf16_kernel<<<nw / 256, 256, 0, stream>>>(Wv, wvb, nw);
    cvt_bf16_kernel<<<nw / 256, 256, 0, stream>>>(Wo, wob, nw);

    ln_bf16_kernel<<<MROWS, 256, 0, stream>>>(x,   lnw, lnb, xn);
    ln_bf16_kernel<<<MROWS, 256, 0, stream>>>(ctx, lnw, lnb, cn);

    dim3 ggrid(DMODEL / 128, MROWS / 128);   // (8, 32)
    gemm_wmma_kernel<false><<<ggrid, 128, 0, stream>>>(xn, wqb, Qb, nullptr, nullptr,
                                                       MROWS, DMODEL, DMODEL);
    gemm_wmma_kernel<false><<<ggrid, 128, 0, stream>>>(cn, wkb, Kb, nullptr, nullptr,
                                                       MROWS, DMODEL, DMODEL);
    gemm_wmma_kernel<false><<<ggrid, 128, 0, stream>>>(cn, wvb, Vb, nullptr, nullptr,
                                                       MROWS, DMODEL, DMODEL);

    flash_attn_kernel<<<dim3(NSEQ / 64, BATCH * HEADS), 128, 0, stream>>>(
        Qb, Kb, Vb, alibi, At);

    gemm_wmma_kernel<true><<<ggrid, 128, 0, stream>>>(At, wob, nullptr, out, bo,
                                                      MROWS, DMODEL, DMODEL);
}
